// DecoderLayer_18408229830863
// MI455X (gfx1250) — matrix operand order repaired and checked by reading, not compiled, not run on hardware
//
#include <hip/hip_runtime.h>

// ---------------------------------------------------------------------------
// MI455X (gfx1250, wave32) fused transformer decoder layer.
// GEMMs: v_wmma_f32_16x16x32_bf16 with double-buffered LDS staging through
// GLOBAL_LOAD_ASYNC_TO_LDS_B128 (ASYNCcnt pipe). Attention: flash-style WMMA.
// ---------------------------------------------------------------------------

#define DIM_  2048
#define NSEQ_ 2048
#define HQn   16
#define HKn   4
#define HDn   128
#define KVn   512
#define FFn   8192

typedef __bf16        v16bf __attribute__((ext_vector_type(16)));
typedef float         v8f   __attribute__((ext_vector_type(8)));
typedef unsigned int  u32x4 __attribute__((ext_vector_type(4)));

union Frag { v16bf bf; u32x4 q[2]; };

__device__ __forceinline__ unsigned short f2bf(float f) {
  unsigned int u = __float_as_uint(f);
  u = (u + 0x7FFFu + ((u >> 16) & 1u)) >> 16;   // round-to-nearest-even
  return (unsigned short)u;
}
__device__ __forceinline__ float bf2f(unsigned short h) {
  return __uint_as_float(((unsigned int)h) << 16);
}

// --------------------------- fp32 -> bf16 conversion -----------------------
__global__ __launch_bounds__(256)
void cvt_f32_bf16(const float* __restrict__ s, unsigned short* __restrict__ d,
                  long long n) {
  long long i = (long long)blockIdx.x * blockDim.x + threadIdx.x;
  long long stride = (long long)gridDim.x * blockDim.x;
  for (; i < n; i += stride) d[i] = f2bf(s[i]);
}

// ------------------------------- RMSNorm -----------------------------------
__global__ __launch_bounds__(256)
void rmsnorm_bf16(const float* __restrict__ x, const float* __restrict__ g,
                  unsigned short* __restrict__ o, int D) {
  int row = blockIdx.x;
  const float* xr = x + (size_t)row * D;
  float s = 0.f;
  for (int i = threadIdx.x; i < D; i += 256) { float v = xr[i]; s += v * v; }
#pragma unroll
  for (int m = 16; m >= 1; m >>= 1) s += __shfl_xor(s, m, 32);
  __shared__ float red[8];
  if ((threadIdx.x & 31) == 0) red[threadIdx.x >> 5] = s;
  __syncthreads();
  float tot = red[0];
#pragma unroll
  for (int w = 1; w < 8; w++) tot += red[w];
  float rs = rsqrtf(tot / (float)D + 1e-6f);
  for (int i = threadIdx.x; i < D; i += 256)
    o[(size_t)row * D + i] = f2bf(xr[i] * rs * g[i]);
}

// ------------------------------- RoPE (+ q scale) --------------------------
__global__ __launch_bounds__(256)
void rope_qk(unsigned short* __restrict__ q, unsigned short* __restrict__ k) {
  const int qPairs = NSEQ_ * HQn * (HDn / 2);
  const int kPairs = NSEQ_ * HKn * (HDn / 2);
  int idx = blockIdx.x * blockDim.x + threadIdx.x;
  if (idx >= qPairs + kPairs) return;
  unsigned short* p; int t, i; float scale;
  if (idx < qPairs) {
    t = idx / (HQn * (HDn / 2));
    int rem = idx % (HQn * (HDn / 2));
    int head = rem / (HDn / 2);
    i = rem % (HDn / 2);
    p = q + (size_t)t * DIM_ + head * HDn + 2 * i;
    scale = 0.08838834764831845f;               // HD^-0.5, folded into q
  } else {
    int id2 = idx - qPairs;
    t = id2 / (HKn * (HDn / 2));
    int rem = id2 % (HKn * (HDn / 2));
    int head = rem / (HDn / 2);
    i = rem % (HDn / 2);
    p = k + (size_t)t * KVn + head * HDn + 2 * i;
    scale = 1.f;
  }
  float inv = powf(10000.f, -(float)(2 * i) / (float)HDn);
  float ang = (float)t * inv;
  float sn, c; sincosf(ang, &sn, &c);
  float xr = bf2f(p[0]), xi = bf2f(p[1]);
  p[0] = f2bf((xr * c - xi * sn) * scale);
  p[1] = f2bf((xr * sn + xi * c) * scale);
}

// ------------------------------- SiLU gate ---------------------------------
__global__ __launch_bounds__(256)
void glu_silu(unsigned short* __restrict__ g1,
              const unsigned short* __restrict__ g3, long long n) {
  long long i = (long long)blockIdx.x * blockDim.x + threadIdx.x;
  long long stride = (long long)gridDim.x * blockDim.x;
  for (; i < n; i += stride) {
    float a = bf2f(g1[i]), b = bf2f(g3[i]);
    float s = a / (1.f + __expf(-a));
    g1[i] = f2bf(s * b);
  }
}

// ------------------------- WMMA GEMM: C = A * B^T --------------------------
// A: MxK bf16 row-major, B: NxK bf16 row-major (torch weight layout).
// Block = 4 waves = 128x128 tile; wave tile 64x64 (4x4 of 16x16 WMMA).
// K loop in steps of 32 with DOUBLE-BUFFERED LDS staging via
// GLOBAL_LOAD_ASYNC_TO_LDS_B128: each thread stages one 64B row-slice of the
// A tile and one of the B tile (4+4 async b128, offset applies to both global
// and LDS address). s_wait_asynccnt 0x8 retires the previous tile while the
// next one remains in flight. Fragments are then ds_load_b128 from LDS.
// LDS rows padded to 80B (40 elems) for bank spread; 2*(10240+10240) = 40KB.
__global__ __launch_bounds__(128)
void gemm_bf16_wmma(const unsigned short* __restrict__ A,
                    const unsigned short* __restrict__ B,
                    const float* __restrict__ bias,
                    const float* __restrict__ resid,
                    float* __restrict__ outF,
                    unsigned short* __restrict__ outH,
                    int M, int N, int K, int transOut) {
  extern __shared__ char smemc[];
  const int t    = threadIdx.x;
  const int lane = t & 31;
  const int wave = t >> 5;
  const int half = lane >> 4;
  const int l15  = lane & 15;
  const int mW   = (wave >> 1) * 64;
  const int nW   = (wave & 1) * 64;
  const int mBlk = blockIdx.y * 128;
  const int nBlk = blockIdx.x * 128;

  const int ROWB   = 80;            // padded row pitch (bytes) = 40 elems
  const int ABYTES = 128 * ROWB;    // 10240
  const int BUFB   = 2 * ABYTES;    // 20480 per buffer (A+B)

  // Per-thread staging sources: one A row and one B row of the block tile.
  unsigned long long gA = (unsigned long long)(A + (size_t)(mBlk + t) * K);
  unsigned long long gB = (unsigned long long)(B + (size_t)(nBlk + t) * K);
  // LDS byte addresses via generic->LDS truncation (flat aperture: addr[31:0]).
  unsigned ldsA0 = (unsigned)(size_t)(smemc + t * ROWB);
  unsigned ldsB0 = (unsigned)(size_t)(smemc + ABYTES + t * ROWB);

  auto stage = [&](int buf, int kElems) {
    unsigned la = ldsA0 + buf * BUFB;
    unsigned lb = ldsB0 + buf * BUFB;
    unsigned long long ga = gA + (unsigned long long)kElems * 2u;
    unsigned long long gb = gB + (unsigned long long)kElems * 2u;
    asm volatile(
        "global_load_async_to_lds_b128 %0, %2, off\n\t"
        "global_load_async_to_lds_b128 %0, %2, off offset:16\n\t"
        "global_load_async_to_lds_b128 %0, %2, off offset:32\n\t"
        "global_load_async_to_lds_b128 %0, %2, off offset:48\n\t"
        "global_load_async_to_lds_b128 %1, %3, off\n\t"
        "global_load_async_to_lds_b128 %1, %3, off offset:16\n\t"
        "global_load_async_to_lds_b128 %1, %3, off offset:32\n\t"
        "global_load_async_to_lds_b128 %1, %3, off offset:48"
        :: "v"(la), "v"(lb), "v"(ga), "v"(gb) : "memory");
  };

  v8f acc[4][4];
#pragma unroll
  for (int i = 0; i < 4; i++)
#pragma unroll
    for (int j = 0; j < 4; j++) acc[i][j] = (v8f){0,0,0,0,0,0,0,0};

  stage(0, 0);                       // prologue: fill buffer 0
  const int steps = K / 32;
  for (int s = 0; s < steps; s++) {
    const int buf = s & 1;
    if (s + 1 < steps) {
      stage(buf ^ 1, (s + 1) * 32);  // prefetch next tile into other buffer
      asm volatile("s_wait_asynccnt 0x8" ::: "memory");  // current tile done
    } else {
      asm volatile("s_wait_asynccnt 0x0" ::: "memory");
    }
    __syncthreads();                 // current tile visible to all waves

    const char* base = smemc + buf * BUFB;
    Frag a[4], b[4];
#pragma unroll
    for (int i = 0; i < 4; i++) {
      // A frag: row mW+16i+l15; 16 consecutive K at byte half*32.
      const u32x4* p = (const u32x4*)(base + (mW + 16*i + l15) * ROWB + half*32);
      a[i].q[0] = p[0];
      a[i].q[1] = p[1];
    }
#pragma unroll
    for (int j = 0; j < 4; j++) {
      // B frag: row nW+16j+l15; 16 consecutive K at byte half*32.
      const u32x4* p = (const u32x4*)(base + ABYTES + (nW + 16*j + l15) * ROWB + half*32);
      b[j].q[0] = p[0];
      b[j].q[1] = p[1];
    }
#pragma unroll
    for (int i = 0; i < 4; i++)
#pragma unroll
      for (int j = 0; j < 4; j++)
        acc[i][j] = __builtin_amdgcn_wmma_f32_16x16x32_bf16(
            false, a[i].bf, false, b[j].bf, (short)0, acc[i][j], false, false);

    __syncthreads();                 // close WAR window before re-staging buf
  }

  // Epilogue. C/D layout: VGPR r <-> row = base + r + 8*half, col = base + l15.
#pragma unroll
  for (int i = 0; i < 4; i++) {
#pragma unroll
    for (int j = 0; j < 4; j++) {
      int col = nBlk + nW + 16*j + l15;
      float bb = bias ? bias[col] : 0.f;
#pragma unroll
      for (int r = 0; r < 8; r++) {
        int row = mBlk + mW + 16*i + half*8 + r;
        float v = acc[i][j][r] + bb;
        if (resid) v += resid[(size_t)row * N + col];
        if (outH) {
          if (transOut) outH[(size_t)col * M + row] = f2bf(v);
          else          outH[(size_t)row * N + col] = f2bf(v);
        } else {
          outF[(size_t)row * N + col] = v;
        }
      }
    }
  }
}

// --------------------------- Flash attention -------------------------------
// grid = (HQ, N/128), block = 256 (8 waves). Each wave: 16 q rows, full HD.
// qr: N x 2048 (roped, pre-scaled), kr: N x 512 (roped), vt: 512 x N (V^T).
// GQA per reference reshape (HQ -> (g, HK)): kv head = h % HK.
// Divergent per-wave trip counts => no block barriers; direct global frags.
__global__ __launch_bounds__(256)
void attn_flash(const unsigned short* __restrict__ qr,
                const unsigned short* __restrict__ kr,
                const unsigned short* __restrict__ vt,
                unsigned short* __restrict__ attn) {
  __shared__ unsigned short pTile[8][16][72];   // 72: pad to avoid bank conflicts
  const int lane = threadIdx.x & 31;
  const int wave = threadIdx.x >> 5;
  const int half = lane >> 4;
  const int l15  = lane & 15;
  const int h    = blockIdx.x;
  const int kh   = h % HKn;
  const int qBase = blockIdx.y * 128 + wave * 16;

  // Q fragments for all of HD=128: 4 chunks of K=32, resident whole kernel.
  Frag qf[4];
  {
    const unsigned short* qrow = qr + (size_t)(qBase + l15) * DIM_ + h * HDn;
#pragma unroll
    for (int c = 0; c < 4; c++) {
      const u32x4* p = (const u32x4*)(qrow + c*32 + half*16);
      qf[c].q[0] = p[0];
      qf[c].q[1] = p[1];
    }
  }

  v8f o[8];
#pragma unroll
  for (int t = 0; t < 8; t++) o[t] = (v8f){0,0,0,0,0,0,0,0};
  float mrun[8], lrun[8];
#pragma unroll
  for (int r = 0; r < 8; r++) { mrun[r] = -3.0e38f; lrun[r] = 0.f; }

  const int jEnd = qBase + 16;                  // causal bound (exclusive)
  for (int jb = 0; jb < jEnd; jb += 64) {
    // ---- S = Q K^T for 16 x 64 tile -------------------------------------
    v8f s[4];
#pragma unroll
    for (int tj = 0; tj < 4; tj++) s[tj] = (v8f){0,0,0,0,0,0,0,0};
#pragma unroll
    for (int c = 0; c < 4; c++) {
      Frag kf[4];
#pragma unroll
      for (int tj = 0; tj < 4; tj++) {
        const u32x4* p = (const u32x4*)(kr + (size_t)(jb + 16*tj + l15) * KVn
                                        + kh * HDn + c*32 + half*16);
        kf[tj].q[0] = p[0];
        kf[tj].q[1] = p[1];
      }
#pragma unroll
      for (int tj = 0; tj < 4; tj++)
        s[tj] = __builtin_amdgcn_wmma_f32_16x16x32_bf16(
            false, qf[c].bf, false, kf[tj].bf, (short)0, s[tj], false, false);
    }

    // ---- online softmax (row stats via 16-lane shfl reductions) ---------
    const bool needMask = (jb + 63) > qBase;
#pragma unroll
    for (int r = 0; r < 8; r++) {
      int row = qBase + r + 8*half;
      float mx = -3.0e38f;
#pragma unroll
      for (int tj = 0; tj < 4; tj++) {
        if (needMask) {
          int j = jb + 16*tj + l15;
          if (j > row) s[tj][r] = -3.0e38f;
        }
        mx = fmaxf(mx, s[tj][r]);
      }
#pragma unroll
      for (int m = 1; m < 16; m <<= 1) mx = fmaxf(mx, __shfl_xor(mx, m, 32));
      float mNew = fmaxf(mrun[r], mx);
      float sc = __expf(mrun[r] - mNew);
      mrun[r] = mNew;
      lrun[r] *= sc;
#pragma unroll
      for (int t = 0; t < 8; t++) o[t][r] *= sc;
      float rs = 0.f;
#pragma unroll
      for (int tj = 0; tj < 4; tj++) {
        float pv = __expf(s[tj][r] - mNew);
        rs += pv;
        pTile[wave][r + 8*half][16*tj + l15] = f2bf(pv);   // P -> LDS (bf16)
      }
#pragma unroll
      for (int m = 1; m < 16; m <<= 1) rs += __shfl_xor(rs, m, 32);
      lrun[r] += rs;
    }
    asm volatile("s_wait_dscnt 0" ::: "memory");

    // ---- O += P * V : A = P from LDS (A-layout), B = V^T (contiguous) ---
    Frag pa[2];
#pragma unroll
    for (int kj = 0; kj < 2; kj++) {
      const u32x4* p = (const u32x4*)(&pTile[wave][l15][kj*32 + half*16]);
      pa[kj].q[0] = p[0];
      pa[kj].q[1] = p[1];
    }
#pragma unroll
    for (int td = 0; td < 8; td++) {
#pragma unroll
      for (int kj = 0; kj < 2; kj++) {
        Frag vb;
        const u32x4* p = (const u32x4*)(vt + (size_t)(kh*HDn + td*16 + l15) * NSEQ_
                                        + jb + kj*32 + half*16);
        vb.q[0] = p[0];
        vb.q[1] = p[1];
        o[td] = __builtin_amdgcn_wmma_f32_16x16x32_bf16(
            false, pa[kj].bf, false, vb.bf, (short)0, o[td], false, false);
      }
    }
  }

  // ---- finalize: divide by l, store bf16 --------------------------------
#pragma unroll
  for (int td = 0; td < 8; td++) {
#pragma unroll
    for (int r = 0; r < 8; r++) {
      int row = qBase + r + 8*half;
      float v = o[td][r] / lrun[r];
      attn[(size_t)row * DIM_ + h * HDn + td*16 + l15] = f2bf(v);
    }
  }
}

// ---------------------------------------------------------------------------
extern "C" void kernel_launch(void* const* d_in, const int* in_sizes, int n_in,
                              void* d_out, int out_size, void* d_ws, size_t ws_size,
                              hipStream_t stream) {
  (void)in_sizes; (void)n_in; (void)out_size; (void)ws_size;
  const float* x  = (const float*)d_in[0];
  const float* wq = (const float*)d_in[1];
  const float* bq = (const float*)d_in[2];
  const float* wk = (const float*)d_in[3];
  const float* bk = (const float*)d_in[4];
  const float* wv = (const float*)d_in[5];
  const float* bv = (const float*)d_in[6];
  const float* wo = (const float*)d_in[7];
  const float* bo = (const float*)d_in[8];
  const float* w1 = (const float*)d_in[9];
  const float* w2 = (const float*)d_in[10];
  const float* w3 = (const float*)d_in[11];
  const float* ga = (const float*)d_in[12];
  const float* gm = (const float*)d_in[13];

  char* ws = (char*)d_ws; size_t off = 0;
  auto alloc = [&](size_t bytes) -> void* {
    void* p = ws + off; off = (off + bytes + 255) & ~(size_t)255; return p;
  };
  typedef unsigned short bf;
  bf* wq_h = (bf*)alloc((size_t)DIM_ * DIM_ * 2);
  bf* wk_h = (bf*)alloc((size_t)KVn  * DIM_ * 2);
  bf* wv_h = (bf*)alloc((size_t)KVn  * DIM_ * 2);
  bf* wo_h = (bf*)alloc((size_t)DIM_ * DIM_ * 2);
  bf* w1_h = (bf*)alloc((size_t)FFn  * DIM_ * 2);
  bf* w2_h = (bf*)alloc((size_t)DIM_ * FFn  * 2);
  bf* w3_h = (bf*)alloc((size_t)FFn  * DIM_ * 2);
  bf* xn   = (bf*)alloc((size_t)NSEQ_ * DIM_ * 2);
  bf* qb   = (bf*)alloc((size_t)NSEQ_ * DIM_ * 2);
  bf* kb   = (bf*)alloc((size_t)NSEQ_ * KVn  * 2);
  bf* vtb  = (bf*)alloc((size_t)KVn  * NSEQ_ * 2);   // V^T (512 x N)
  bf* atb  = (bf*)alloc((size_t)NSEQ_ * DIM_ * 2);
  float* hb = (float*)alloc((size_t)NSEQ_ * DIM_ * 4);
  bf* hn   = (bf*)alloc((size_t)NSEQ_ * DIM_ * 2);
  bf* g1   = (bf*)alloc((size_t)NSEQ_ * FFn * 2);
  bf* g3   = (bf*)alloc((size_t)NSEQ_ * FFn * 2);

  auto cvt = [&](const float* s, bf* d, long long n) {
    int blocks = (int)((n + 256LL*8 - 1) / (256LL*8));
    if (blocks > 4096) blocks = 4096;
    cvt_f32_bf16<<<blocks, 256, 0, stream>>>(s, d, n);
  };
  cvt(wq, wq_h, (long long)DIM_*DIM_);
  cvt(wk, wk_h, (long long)KVn*DIM_);
  cvt(wv, wv_h, (long long)KVn*DIM_);
  cvt(wo, wo_h, (long long)DIM_*DIM_);
  cvt(w1, w1_h, (long long)FFn*DIM_);
  cvt(w2, w2_h, (long long)DIM_*FFn);
  cvt(w3, w3_h, (long long)FFn*DIM_);

  const int GEMM_LDS = 2 * 2 * 128 * 80;   // 40960 bytes
  auto gemm = [&](const bf* A, const bf* Bm, const float* bias, const float* resid,
                  float* oF, bf* oH, int M, int Nn, int K, int tr) {
    dim3 g(Nn / 128, M / 128);
    gemm_bf16_wmma<<<g, 128, GEMM_LDS, stream>>>(A, Bm, bias, resid, oF, oH, M, Nn, K, tr);
  };

  // 1) xn = rmsnorm(x) (bf16)
  rmsnorm_bf16<<<NSEQ_, 256, 0, stream>>>(x, ga, xn, DIM_);
  // 2) projections (bias fused); V stored transposed for attention B-frags
  gemm(xn, wq_h, bq, nullptr, nullptr, qb, NSEQ_, DIM_, DIM_, 0);
  gemm(xn, wk_h, bk, nullptr, nullptr, kb, NSEQ_, KVn,  DIM_, 0);
  gemm(xn, wv_h, bv, nullptr, nullptr, vtb, NSEQ_, KVn, DIM_, 1);
  // 3) RoPE on q (with HD^-0.5) and k
  {
    long long tot = (long long)NSEQ_*HQn*(HDn/2) + (long long)NSEQ_*HKn*(HDn/2);
    int blocks = (int)((tot + 255) / 256);
    rope_qk<<<blocks, 256, 0, stream>>>(qb, kb);
  }
  // 4) flash attention
  {
    dim3 g(HQn, NSEQ_ / 128);
    attn_flash<<<g, 256, 0, stream>>>(qb, kb, vtb, atb);
  }
  // 5) h = x + attn @ wo^T + bo   (fp32)
  gemm(atb, wo_h, bo, x, hb, nullptr, NSEQ_, DIM_, DIM_, 0);
  // 6) hn = rmsnorm(h)
  rmsnorm_bf16<<<NSEQ_, 256, 0, stream>>>(hb, gm, hn, DIM_);
  // 7) g1 = hn @ w1^T ; g3 = hn @ w3^T
  gemm(hn, w1_h, nullptr, nullptr, nullptr, g1, NSEQ_, FFn, DIM_, 0);
  gemm(hn, w3_h, nullptr, nullptr, nullptr, g3, NSEQ_, FFn, DIM_, 0);
  // 8) g1 = silu(g1) * g3
  {
    long long tot = (long long)NSEQ_ * FFn;
    int blocks = (int)((tot + 256LL*8 - 1) / (256LL*8));
    if (blocks > 4096) blocks = 4096;
    glu_silu<<<blocks, 256, 0, stream>>>(g1, g3, tot);
  }
  // 9) out = h + g1 @ w2^T   (fp32, final output)
  gemm(g1, w2_h, nullptr, hb, (float*)d_out, nullptr, NSEQ_, DIM_, FFn, 0);
}
